// HybridRecursiveStateSpace_31748398252144
// MI455X (gfx1250) — compile-verified, matrix operand order checked by reading, not run on hardware
//
#include <hip/hip_runtime.h>
#include <hip/hip_bf16.h>
#include <math.h>

// ---------------- model constants ----------------
#define BATCH   2
#define SEQ     448
#define DMODEL  512
#define DINNER  1024
#define DSTATE  16
#define DCONV   4
#define NHEADS  8
#define DHEAD   64
#define VOCAB   50257
#define ROWS    (BATCH*SEQ)          // 896

typedef __bf16 bf16;
typedef __attribute__((ext_vector_type(16))) __bf16 v16bf;
typedef __attribute__((ext_vector_type(8)))  __bf16 v8bf;
typedef __attribute__((ext_vector_type(8)))  float  v8f;

__device__ __forceinline__ float siluf(float x) { return x / (1.0f + __expf(-x)); }
__device__ __forceinline__ float geluf(float x) { return 0.5f * x * (1.0f + erff(x * 0.70710678118654752f)); }
__device__ __forceinline__ float softplusf(float x) { return (x > 20.0f) ? x : log1pf(__expf(x)); }

// =====================================================================
// Fused WMMA GEMM:  C = act(A[M,K] * B^T + bias) + R   (f32 in/out, bf16
// convert while staging to LDS).  B is [N,K] row-major (weights layout).
// Double-buffered LDS + register prefetch: global loads for tile k+1 are
// issued before the WMMAs of tile k so their latency hides behind matrix
// math; one barrier per K-step.  Single staging path (no runtime layout
// branch) keeps codegen clean — transposed operands are materialized by a
// tiny transpose kernel instead.
// 2-level batching: z -> (zb, zh) = (z/bh, z%bh) with separate strides.
// M multiple of 64, K multiple of 32 (true for every call); N is guarded.
// =====================================================================
__launch_bounds__(128)
__global__ void gemm_wmma(const float* __restrict__ A, const float* __restrict__ B,
                          const float* __restrict__ bias, const float* __restrict__ R,
                          float* __restrict__ C,
                          int M, int N, int K, int lda, int ldb, int ldc,
                          int bh,
                          long long sAb, long long sAh,
                          long long sBb, long long sBh,
                          long long sCb, long long sCh,
                          int act)
{
    __shared__ bf16 As[2][64][32];
    __shared__ bf16 Bs[2][64][32];

    const int z  = blockIdx.z;
    const int zb = z / bh, zh = z % bh;
    A += (size_t)zb * sAb + (size_t)zh * sAh;
    B += (size_t)zb * sBb + (size_t)zh * sBh;
    C += (size_t)zb * sCb + (size_t)zh * sCh;
    if (R) R += (size_t)zb * sCb + (size_t)zh * sCh;   // R always shares C's layout

    const int m0  = blockIdx.y * 64;
    const int n0  = blockIdx.x * 64;
    const int tid = threadIdx.x;           // 0..127
    const int wave = tid >> 5;             // 0..3
    const int lane = tid & 31;
    const int wm = (wave >> 1) * 32;       // wave row sub-tile
    const int wn = (wave & 1)  * 32;       // wave col sub-tile
    const int lm  = lane & 15;
    const int lhi = lane >> 4;             // 0/1

    // staging coordinates (constant per thread)
    const int ar = tid >> 3;               // 0..15  (row base; +16 per chunk)
    const int ak = (tid & 7) << 2;         // 0,4,..,28
    const bool tileFullN = (n0 + 64 <= N);

    v8f acc[2][2];
#pragma unroll
    for (int i = 0; i < 2; ++i)
#pragma unroll
        for (int j = 0; j < 2; ++j) acc[i][j] = (v8f){0,0,0,0,0,0,0,0};

    float4 fa[4];
    float4 fb[4];

    // ---- prefetch helpers (unrolled, independent loads) ----
    auto loadA = [&](int k0) {
#pragma unroll
        for (int u = 0; u < 4; ++u)
            fa[u] = *(const float4*)(A + (size_t)(m0 + ar + u * 16) * lda + k0 + ak);
    };
    auto loadB = [&](int k0) {
        if (tileFullN) {
#pragma unroll
            for (int u = 0; u < 4; ++u)
                fb[u] = *(const float4*)(B + (size_t)(n0 + ar + u * 16) * ldb + k0 + ak);
        } else {
#pragma unroll
            for (int u = 0; u < 4; ++u) {
                int n = n0 + ar + u * 16;
                fb[u] = make_float4(0.f, 0.f, 0.f, 0.f);
                if (n < N)
                    fb[u] = *(const float4*)(B + (size_t)n * ldb + k0 + ak);
            }
        }
    };
    auto stage = [&](int s) {
#pragma unroll
        for (int u = 0; u < 4; ++u) {
            As[s][ar + u * 16][ak + 0] = (bf16)fa[u].x;
            As[s][ar + u * 16][ak + 1] = (bf16)fa[u].y;
            As[s][ar + u * 16][ak + 2] = (bf16)fa[u].z;
            As[s][ar + u * 16][ak + 3] = (bf16)fa[u].w;
        }
#pragma unroll
        for (int u = 0; u < 4; ++u) {
            Bs[s][ar + u * 16][ak + 0] = (bf16)fb[u].x;
            Bs[s][ar + u * 16][ak + 1] = (bf16)fb[u].y;
            Bs[s][ar + u * 16][ak + 2] = (bf16)fb[u].z;
            Bs[s][ar + u * 16][ak + 3] = (bf16)fb[u].w;
        }
    };

    // ---- prologue: stage tile 0 into buffer 0 ----
    loadA(0);
    loadB(0);
    stage(0);
    __syncthreads();

    const int nk = K >> 5;
    for (int it = 0; it < nk; ++it) {
        const int cur  = it & 1;
        const int nxt  = cur ^ 1;
        const bool more = (it + 1 < nk);

        // issue next tile's global loads before touching LDS
        if (more) {
            loadA((it + 1) << 5);
            loadB((it + 1) << 5);
        }

        // ---- fragments from LDS[cur] (two b128 per lane per fragment) ----
        v16bf af[2], bfv[2];
#pragma unroll
        for (int i = 0; i < 2; ++i) {
            union { v16bf v; v8bf h[2]; } ua, ub;
            int mrow = wm + i * 16 + lm;
            ua.h[0] = *(const v8bf*)&As[cur][mrow][lhi * 8];
            ua.h[1] = *(const v8bf*)&As[cur][mrow][16 + lhi * 8];
            af[i] = ua.v;
            int nrow = wn + i * 16 + lm;
            ub.h[0] = *(const v8bf*)&Bs[cur][nrow][lhi * 8];
            ub.h[1] = *(const v8bf*)&Bs[cur][nrow][16 + lhi * 8];
            bfv[i] = ub.v;
        }
#pragma unroll
        for (int i = 0; i < 2; ++i)
#pragma unroll
            for (int j = 0; j < 2; ++j)
                acc[i][j] = __builtin_amdgcn_wmma_f32_16x16x32_bf16(
                    false, af[i], false, bfv[j], (short)0, acc[i][j], false, false);

        if (more) {
            stage(nxt);          // convert + ds_store (waits for the global loads here)
            __syncthreads();     // one barrier per K-step
        }
    }

    // ---- epilogue: bias + activation + residual, f32 store ----
#pragma unroll
    for (int i = 0; i < 2; ++i)
#pragma unroll
        for (int j = 0; j < 2; ++j) {
            int col = n0 + wn + j * 16 + lm;
            if (col >= N) continue;
            float bv = bias ? bias[col] : 0.f;
#pragma unroll
            for (int r = 0; r < 8; ++r) {
                int row = m0 + wm + i * 16 + r + lhi * 8;
                float v = acc[i][j][r] + bv;
                if (act == 1)      v = geluf(v);
                else if (act == 2) v = siluf(v);
                if (R) v += R[(size_t)row * ldc + col];
                C[(size_t)row * ldc + col] = v;
            }
        }
}

// =====================================================================
// Elementwise / reduction kernels
// =====================================================================
__global__ void embed_kernel(const int* __restrict__ ids, const float* __restrict__ emb,
                             const float* __restrict__ pos, float* __restrict__ x)
{
    int idx = blockIdx.x * 256 + threadIdx.x;
    if (idx >= ROWS * DMODEL) return;
    int d = idx % DMODEL;
    int m = idx / DMODEL;
    int l = m % SEQ;
    x[idx] = emb[(size_t)ids[m] * DMODEL + d] + pos[(size_t)l * DMODEL + d];
}

__launch_bounds__(256)
__global__ void ln_kernel(const float* __restrict__ x, const float* __restrict__ w,
                          const float* __restrict__ b, float* __restrict__ out)
{
    int row = blockIdx.x;
    const float* xr = x + (size_t)row * DMODEL;
    float* orow = out + (size_t)row * DMODEL;
    int tid = threadIdx.x;                    // 256 threads, D=512 -> 2 elems each
    float v0 = xr[tid], v1 = xr[tid + 256];
    __shared__ float r1[256], r2[256];
    r1[tid] = v0 + v1;
    r2[tid] = v0 * v0 + v1 * v1;
    __syncthreads();
    for (int s = 128; s > 0; s >>= 1) {
        if (tid < s) { r1[tid] += r1[tid + s]; r2[tid] += r2[tid + s]; }
        __syncthreads();
    }
    float mean = r1[0] * (1.0f / DMODEL);
    float var  = r2[0] * (1.0f / DMODEL) - mean * mean;
    float inv  = rsqrtf(var + 1e-5f);
    orow[tid]       = (v0 - mean) * inv * w[tid]       + b[tid];
    orow[tid + 256] = (v1 - mean) * inv * w[tid + 256] + b[tid + 256];
}

// depthwise causal conv1d (+bias) + SiLU; xi = first DINNER cols of xz
__global__ void conv_silu_kernel(const float* __restrict__ xz, const float* __restrict__ w,
                                 const float* __restrict__ cb, float* __restrict__ xc)
{
    int idx = blockIdx.x * 256 + threadIdx.x;
    if (idx >= ROWS * DINNER) return;
    int d = idx % DINNER;
    int m = idx / DINNER;
    int l = m % SEQ, b = m / SEQ;
    float acc = cb[d];
#pragma unroll
    for (int j = 0; j < DCONV; ++j) {
        int ls = l - (DCONV - 1) + j;
        if (ls >= 0)
            acc += w[d * DCONV + j] * xz[(size_t)(b * SEQ + ls) * (2 * DINNER) + d];
    }
    xc[idx] = siluf(acc);
}

// delta[m,d] = softplus(xs[m,0] * dt_w[d] + dt_b[d])
__global__ void delta_kernel(const float* __restrict__ xs, const float* __restrict__ dtw,
                             const float* __restrict__ dtb, float* __restrict__ delta)
{
    int idx = blockIdx.x * 256 + threadIdx.x;
    if (idx >= ROWS * DINNER) return;
    int d = idx % DINNER;
    int m = idx / DINNER;
    delta[idx] = softplusf(xs[(size_t)m * 33] * dtw[d] + dtb[d]);
}

// sequential selective-scan: one lane per (batch, channel)
__global__ void scan_kernel(const float* __restrict__ xs, const float* __restrict__ delta,
                            const float* __restrict__ xc, const float* __restrict__ Alog,
                            float* __restrict__ y)
{
    int t = blockIdx.x * blockDim.x + threadIdx.x;
    if (t >= BATCH * DINNER) return;
    int b = t / DINNER, d = t % DINNER;
    float Ad[DSTATE];
#pragma unroll
    for (int s = 0; s < DSTATE; ++s) Ad[s] = -__expf(Alog[(size_t)d * DSTATE + s]);
    float h[DSTATE];
#pragma unroll
    for (int s = 0; s < DSTATE; ++s) h[s] = 0.f;

    for (int l = 0; l < SEQ; ++l) {
        size_t m = (size_t)(b * SEQ + l);
        float dt = delta[m * DINNER + d];
        float xv = xc[m * DINNER + d];
        const float* xrow = xs + m * 33;
        float acc = 0.f;
#pragma unroll
        for (int s = 0; s < DSTATE; ++s) {
            float dA = __expf(dt * Ad[s]);
            h[s] = dA * h[s] + dt * xrow[1 + s] * xv;
            acc += h[s] * xrow[17 + s];
        }
        y[m * DINNER + d] = acc;
    }
}

// y = (y + xc * D[d]) * silu(z);  z = second DINNER cols of xz
__global__ void gate_kernel(float* __restrict__ y, const float* __restrict__ xc,
                            const float* __restrict__ xz, const float* __restrict__ Dp)
{
    int idx = blockIdx.x * 256 + threadIdx.x;
    if (idx >= ROWS * DINNER) return;
    int d = idx % DINNER;
    int m = idx / DINNER;
    float zv = xz[(size_t)m * (2 * DINNER) + DINNER + d];
    y[idx] = (y[idx] + xc[idx] * Dp[d]) * siluf(zv);
}

// causal softmax over one row of S (with 1/sqrt(Dh) scale), in place
__launch_bounds__(256)
__global__ void softmax_causal_kernel(float* __restrict__ S)
{
    int row = blockIdx.x;                 // (b*H + h)*SEQ + q
    int q = row % SEQ;
    float* sr = S + (size_t)row * SEQ;
    int tid = threadIdx.x;
    __shared__ float red[256];
    const float scale = 0.125f;           // DHEAD^-0.5

    float mx = -1e30f;
    for (int k = tid; k <= q; k += 256) mx = fmaxf(mx, sr[k] * scale);
    red[tid] = mx;  __syncthreads();
    for (int s = 128; s > 0; s >>= 1) {
        if (tid < s) red[tid] = fmaxf(red[tid], red[tid + s]);
        __syncthreads();
    }
    mx = red[0];
    __syncthreads();

    float sum = 0.f;
    for (int k = tid; k < SEQ; k += 256) {
        float v = (k <= q) ? __expf(sr[k] * scale - mx) : 0.f;
        sr[k] = v;
        sum += v;
    }
    red[tid] = sum;  __syncthreads();
    for (int s = 128; s > 0; s >>= 1) {
        if (tid < s) red[tid] += red[tid + s];
        __syncthreads();
    }
    float inv = 1.f / red[0];
    for (int k = tid; k < SEQ; k += 256) sr[k] *= inv;
}

// materialize V^T per head: Vt[(b*H+h)][n][k] = qkv[b*SEQ + k][2*DMODEL + h*DHEAD + n]
// (1.8 MB one-shot copy so P@V can use the fast row-major GEMM staging path)
__global__ void vtrans_kernel(const float* __restrict__ qkv, float* __restrict__ Vt)
{
    int idx = blockIdx.x * 256 + threadIdx.x;
    if (idx >= BATCH * NHEADS * DHEAD * SEQ) return;
    int k  = idx % SEQ;
    int r  = idx / SEQ;
    int n  = r % DHEAD;
    int bh = r / DHEAD;
    int b = bh / NHEADS, h = bh % NHEADS;
    Vt[idx] = qkv[(size_t)(b * SEQ + k) * (3 * DMODEL) + 2 * DMODEL + h * DHEAD + n];
}

// =====================================================================
// Host orchestration
// =====================================================================
extern "C" void kernel_launch(void* const* d_in, const int* in_sizes, int n_in,
                              void* d_out, int out_size, void* d_ws, size_t ws_size,
                              hipStream_t stream)
{
    (void)in_sizes; (void)n_in; (void)out_size; (void)ws_size;

    // ---- input indexing (setup_inputs dict order) ----
    auto F = [&](int i) { return (const float*)d_in[i]; };
    int cur = 0;
    const int* ids = (const int*)d_in[cur++];       // input_ids
    const float* emb = F(cur++);                    // embedding
    const float* pos = F(cur++);                    // pos_embedding
    const char types[5] = {'m', 'm', 'a', 'm', 'm'};
    int blk[5];
    for (int i = 0; i < 5; ++i) { blk[i] = cur; cur += (types[i] == 'm') ? 11 : 10; }
    int adp[2];
    adp[0] = cur; cur += 4;
    adp[1] = cur; cur += 4;
    const float* norm_w = F(cur++);
    const float* norm_b = F(cur++);
    const float* lm_head = F(cur++);

    // ---- workspace layout (all multiples of 4 floats -> float4 aligned) ----
    float* p = (float*)d_ws;
    float* xA   = p; p += ROWS * DMODEL;
    float* xB   = p; p += ROWS * DMODEL;
    float* hbuf = p; p += ROWS * DMODEL;
    float* xz   = p; p += ROWS * 2 * DINNER;
    float* xc   = p; p += ROWS * DINNER;
    float* xs   = p; p += ROWS * 33 + 3;            // keep next buffer 16B aligned
    p = (float*)(((uintptr_t)p + 15) & ~(uintptr_t)15);
    float* dlt  = p; p += ROWS * DINNER;
    float* yy   = p; p += ROWS * DINNER;
    float* qkv  = p; p += ROWS * 3 * DMODEL;
    float* Sbuf = p; p += (size_t)BATCH * NHEADS * SEQ * SEQ;
    float* obuf = p; p += ROWS * DMODEL;
    float* fbuf = p; p += ROWS * 4 * DMODEL;
    float* Vt   = p; p += (size_t)BATCH * NHEADS * DHEAD * SEQ;

    float* xcur = xA;
    float* xnxt = xB;

    auto gemm = [&](const float* A, const float* B, const float* bias, const float* R,
                    float* C, int M, int N, int K, int lda, int ldb, int ldc,
                    int batch, int bh,
                    long long sAb, long long sAh, long long sBb, long long sBh,
                    long long sCb, long long sCh, int act) {
        dim3 g((N + 63) / 64, M / 64, batch);
        gemm_wmma<<<g, 128, 0, stream>>>(A, B, bias, R, C, M, N, K, lda, ldb, ldc,
                                         bh, sAb, sAh, sBb, sBh, sCb, sCh, act);
    };
    auto gemm1 = [&](const float* A, const float* B, const float* bias, const float* R,
                     float* C, int M, int N, int K, int act) {
        gemm(A, B, bias, R, C, M, N, K, K, K, N, 1, 1, 0, 0, 0, 0, 0, 0, act);
    };
    auto ln = [&](const float* x, const float* w, const float* b, float* out) {
        ln_kernel<<<ROWS, 256, 0, stream>>>(x, w, b, out);
    };

    // ---- embedding + positional ----
    embed_kernel<<<(ROWS * DMODEL + 255) / 256, 256, 0, stream>>>(ids, emb, pos, xcur);

    const int nEl = ROWS * DINNER;
    for (int depth = 0; depth < 2; ++depth) {
        for (int bi = 0; bi < 5; ++bi) {
            int pidx = blk[bi];
            if (types[bi] == 'm') {
                // ---- Mamba block ----
                ln(xcur, F(pidx + 0), F(pidx + 1), hbuf);
                gemm1(hbuf, F(pidx + 2), nullptr, nullptr, xz,
                      ROWS, 2 * DINNER, DMODEL, 0);                       // in_proj
                conv_silu_kernel<<<(nEl + 255) / 256, 256, 0, stream>>>(
                    xz, F(pidx + 3), F(pidx + 4), xc);
                gemm1(xc, F(pidx + 5), nullptr, nullptr, xs,
                      ROWS, 33, DINNER, 0);                                // x_proj
                delta_kernel<<<(nEl + 255) / 256, 256, 0, stream>>>(
                    xs, F(pidx + 6), F(pidx + 7), dlt);
                scan_kernel<<<(BATCH * DINNER + 255) / 256, 256, 0, stream>>>(
                    xs, dlt, xc, F(pidx + 8), yy);
                gate_kernel<<<(nEl + 255) / 256, 256, 0, stream>>>(
                    yy, xc, xz, F(pidx + 9));
                gemm1(yy, F(pidx + 10), nullptr, xcur, xnxt,
                      ROWS, DMODEL, DINNER, 0);                            // out_proj + residual
                { float* t = xcur; xcur = xnxt; xnxt = t; }
            } else {
                // ---- Attention block ----
                ln(xcur, F(pidx + 0), F(pidx + 1), hbuf);
                gemm1(hbuf, F(pidx + 2), nullptr, nullptr, qkv,
                      ROWS, 3 * DMODEL, DMODEL, 0);                        // qkv
                // S = Q @ K^T, batched over (b, h)
                gemm(qkv + 0 * DMODEL, qkv + 1 * DMODEL, nullptr, nullptr, Sbuf,
                     SEQ, SEQ, DHEAD, 3 * DMODEL, 3 * DMODEL, SEQ,
                     BATCH * NHEADS, NHEADS,
                     (long long)SEQ * 3 * DMODEL, DHEAD,
                     (long long)SEQ * 3 * DMODEL, DHEAD,
                     (long long)NHEADS * SEQ * SEQ, (long long)SEQ * SEQ, 0);
                softmax_causal_kernel<<<BATCH * NHEADS * SEQ, 256, 0, stream>>>(Sbuf);
                // materialize V^T per head, then O = P @ (V^T)^T via fast path
                vtrans_kernel<<<(BATCH * NHEADS * DHEAD * SEQ + 255) / 256, 256, 0, stream>>>(
                    qkv, Vt);
                gemm(Sbuf, Vt, nullptr, nullptr, obuf,
                     SEQ, DHEAD, SEQ, SEQ, SEQ, DMODEL,
                     BATCH * NHEADS, NHEADS,
                     (long long)NHEADS * SEQ * SEQ, (long long)SEQ * SEQ,
                     (long long)NHEADS * DHEAD * SEQ, (long long)DHEAD * SEQ,
                     (long long)SEQ * DMODEL, DHEAD, 0);
                gemm1(obuf, F(pidx + 3), nullptr, xcur, xnxt,
                      ROWS, DMODEL, DMODEL, 0);                            // out + residual
                { float* t = xcur; xcur = xnxt; xnxt = t; }
                // MLP
                ln(xcur, F(pidx + 4), F(pidx + 5), hbuf);
                gemm1(hbuf, F(pidx + 6), F(pidx + 7), nullptr, fbuf,
                      ROWS, 4 * DMODEL, DMODEL, 1);                        // w1 + b1 + gelu
                gemm1(fbuf, F(pidx + 8), F(pidx + 9), xcur, xnxt,
                      ROWS, DMODEL, 4 * DMODEL, 0);                        // w2 + b2 + residual
                { float* t = xcur; xcur = xnxt; xnxt = t; }
            }
        }
        // ---- adapter: x = x + gelu(ln(x) @ w^T + b) ----
        int a = adp[depth];
        ln(xcur, F(a + 0), F(a + 1), hbuf);
        gemm1(hbuf, F(a + 2), F(a + 3), xcur, xnxt, ROWS, DMODEL, DMODEL, 1);
        { float* t = xcur; xcur = xnxt; xnxt = t; }
    }

    // ---- final LN + lm_head ----
    ln(xcur, norm_w, norm_b, hbuf);
    gemm1(hbuf, lm_head, nullptr, nullptr, (float*)d_out, ROWS, VOCAB, DMODEL, 0);
}
